// MultiHeadAttention_35665408426743
// MI455X (gfx1250) — compile-verified, matrix operand order checked by reading, not run on hardware
//
#include <hip/hip_runtime.h>

#define B_ 4
#define S_ 2048
#define E_ 1024
#define H_ 16
#define D_ 64

typedef __bf16 bf16;
typedef __attribute__((ext_vector_type(16))) bf16 v16bf;
typedef __attribute__((ext_vector_type(8)))  bf16 v8bf;
typedef __attribute__((ext_vector_type(8)))  float v8f;
typedef int v4i __attribute__((vector_size(16)));   // matches builtin proto
typedef __attribute__((address_space(3))) bf16 las_bf16;

__device__ __forceinline__ v8f wmma_bf16(v16bf a, v16bf b, v8f c) {
  return __builtin_amdgcn_wmma_f32_16x16x32_bf16(
      /*neg_a=*/false, a, /*neg_b=*/false, b,
      /*c_mod=*/(short)0, c, /*reuse_a=*/false, /*reuse_b=*/false);
}

// 16-byte async global->LDS copy (per-lane addresses, tracked by ASYNCcnt)
__device__ __forceinline__ void async_copy_b128(const bf16* g, bf16* l) {
#if __has_builtin(__builtin_amdgcn_global_load_async_to_lds_b128)
  __builtin_amdgcn_global_load_async_to_lds_b128((v4i*)g, (v4i*)l, 0, 0);
#else
  unsigned lds_off = (unsigned)(__UINTPTR_TYPE__)(las_bf16*)l;
  unsigned long long ga = (unsigned long long)(__UINTPTR_TYPE__)g;
  asm volatile("global_load_async_to_lds_b128 %0, %1, off"
               :: "v"(lds_off), "v"(ga) : "memory");
#endif
}

__device__ __forceinline__ void wait_async0() {
#if __has_builtin(__builtin_amdgcn_s_wait_asynccnt)
  __builtin_amdgcn_s_wait_asynccnt(0);
#else
  asm volatile("s_wait_asynccnt 0x0" ::: "memory");
#endif
}

// A/B fragment load from K-contiguous row-major storage.
// p = row_start + k_base + 8*half.  Per ISA: VGPR0-3 hold K={kb..kb+7},
// VGPR4-7 hold K={kb+16..kb+23} (kb = 8*half per lane-half).
__device__ __forceinline__ v16bf load_frag_bf16(const bf16* p) {
  v8bf lo = *(const v8bf*)(p);
  v8bf hi = *(const v8bf*)(p + 16);
  v16bf r;
#pragma unroll
  for (int i = 0; i < 8; ++i) { r[i] = lo[i]; r[8 + i] = hi[i]; }
  return r;
}

__device__ __forceinline__ v16bf load_frag_f32(const float* p) {
  float4 a0 = *(const float4*)(p);
  float4 a1 = *(const float4*)(p + 4);
  float4 b0 = *(const float4*)(p + 16);
  float4 b1 = *(const float4*)(p + 20);
  v16bf r;
  r[0]  = (bf16)a0.x; r[1]  = (bf16)a0.y; r[2]  = (bf16)a0.z; r[3]  = (bf16)a0.w;
  r[4]  = (bf16)a1.x; r[5]  = (bf16)a1.y; r[6]  = (bf16)a1.z; r[7]  = (bf16)a1.w;
  r[8]  = (bf16)b0.x; r[9]  = (bf16)b0.y; r[10] = (bf16)b0.z; r[11] = (bf16)b0.w;
  r[12] = (bf16)b1.x; r[13] = (bf16)b1.y; r[14] = (bf16)b1.z; r[15] = (bf16)b1.w;
  return r;
}

// ---------------- Kernel 0: fp32 -> bf16 batched transpose ----------------
__global__ __launch_bounds__(256) void cvt_transpose_kernel(
    const float* __restrict__ W, bf16* __restrict__ WT, int nB, int R, int C) {
  long long idx = (long long)blockIdx.x * 256 + threadIdx.x;
  long long total = (long long)nB * R * C;
  if (idx >= total) return;
  int c = (int)(idx % C);
  long long t = idx / C;
  int r = (int)(t % R);
  int bi = (int)(t / R);
  WT[((long long)bi * C + c) * R + r] = (bf16)W[idx];
}

// ---------------- Kernel 1: per-head projection GEMM ----------------
template <int TRANSPOSED>
__global__ __launch_bounds__(256) void proj_kernel(
    const float* __restrict__ X, const bf16* __restrict__ WT,
    const float* __restrict__ bias, bf16* __restrict__ out, float outScale) {
  const int bh = blockIdx.y;
  const int b = bh / H_, h = bh % H_;
  const int wave = threadIdx.x >> 5;
  const int lane = threadIdx.x & 31;
  const int half = lane >> 4;
  const int ln   = lane & 15;
  const int rowBase = blockIdx.x * 128 + wave * 16;

  v8f c[4];
#pragma unroll
  for (int f = 0; f < 4; ++f)
#pragma unroll
    for (int j = 0; j < 8; ++j) c[f][j] = 0.f;

  const float* xrow  = X + ((size_t)b * S_ + rowBase + ln) * E_;
  const bf16*  wbase = WT + (size_t)h * D_ * E_ + (size_t)ln * E_;

  for (int e = 0; e < E_; e += 32) {
    if (e + 128 < E_) __builtin_prefetch(xrow + e + 128, 0, 0);
    v16bf wf[4];
#pragma unroll
    for (int dt = 0; dt < 4; ++dt)
      wf[dt] = load_frag_bf16(wbase + (size_t)(dt * 16) * E_ + e + 8 * half);
    v16bf a = load_frag_f32(xrow + e + 8 * half);
#pragma unroll
    for (int dt = 0; dt < 4; ++dt) c[dt] = wmma_bf16(a, wf[dt], c[dt]);
  }

#pragma unroll
  for (int dt = 0; dt < 4; ++dt) {
    const int d = dt * 16 + ln;
    const float bv = bias[h * D_ + d];
    if (TRANSPOSED) {
      v8bf pk;
#pragma unroll
      for (int j = 0; j < 8; ++j) pk[j] = (bf16)((c[dt][j] + bv) * outScale);
      *(v8bf*)(out + ((size_t)bh * D_ + d) * S_ + rowBase + 8 * half) = pk;
    } else {
#pragma unroll
      for (int j = 0; j < 8; ++j) {
        int s = rowBase + j + 8 * half;
        out[((size_t)bh * S_ + s) * D_ + d] = (bf16)((c[dt][j] + bv) * outScale);
      }
    }
  }
}

// ---------------- Kernel 2: flash attention per (b,h) ----------------
// Q [B*H,S,D] bf16 (pre-scaled by (1/sqrt(D))*log2e) ; K [B*H,S,D] bf16 ;
// VT [B*H,D,S] bf16 ; X [B,S,E] bf16.
// 4 waves x 16 rows; t-block = 64 keys; K/V tiles double-buffered in LDS via
// async global->LDS copies (ASYNCcnt), staged cooperatively by the block.
__global__ __launch_bounds__(128) void attn_kernel(
    const bf16* __restrict__ Q, const bf16* __restrict__ K,
    const bf16* __restrict__ VT, bf16* __restrict__ X) {
  __shared__ bf16 kbuf[2][64][64];    // K rows [t][d]      (8KB each)
  __shared__ bf16 vbuf[2][64][64];    // V^T rows [d][t]    (8KB each)
  __shared__ bf16 pstage[4][16][64];  // per-wave P relayout (2KB each)

  const int tid  = threadIdx.x;
  const int bh   = blockIdx.y;
  const int b = bh / H_, h = bh % H_;
  const int wave = tid >> 5;
  const int lane = tid & 31;
  const int half = lane >> 4;
  const int ln   = lane & 15;
  const int rowBase = blockIdx.x * 64 + wave * 16;

  const bf16* kbase = K  + (size_t)bh * S_ * D_;
  const bf16* vbase = VT + (size_t)bh * D_ * S_;

  // cooperative async stage of one 64-key block: 512 16B chunks of K + V
  auto stage_kv = [&](int t0, int buf) {
#pragma unroll
    for (int i = 0; i < 4; ++i) {
      int c = tid + i * 128;                  // 0..511
      int row = c >> 3, col = (c & 7) * 8;    // 64 rows x 64 elems
      async_copy_b128(kbase + (size_t)(t0 + row) * D_ + col, &kbuf[buf][row][col]);
    }
#pragma unroll
    for (int i = 0; i < 4; ++i) {
      int c = tid + i * 128;
      int row = c >> 3, col = (c & 7) * 8;
      async_copy_b128(vbase + (size_t)row * S_ + t0 + col, &vbuf[buf][row][col]);
    }
  };

  const bf16* qrow = Q + ((size_t)bh * S_ + rowBase + ln) * D_;
  const v16bf qa0 = load_frag_bf16(qrow + 0  + 8 * half);   // d = 0..31
  const v16bf qa1 = load_frag_bf16(qrow + 32 + 8 * half);   // d = 32..63

  v8f o[4];
  float m[8], l[8];
#pragma unroll
  for (int f = 0; f < 4; ++f)
#pragma unroll
    for (int j = 0; j < 8; ++j) o[f][j] = 0.f;
#pragma unroll
  for (int j = 0; j < 8; ++j) { m[j] = -1e30f; l[j] = 0.f; }

  stage_kv(0, 0);

  for (int t0 = 0; t0 < S_; t0 += 64) {
    const int cur = (t0 >> 6) & 1;
    wait_async0();      // this wave's staged chunks landed
    __syncthreads();    // every wave's chunks landed; prior readers done
    if (t0 + 64 < S_) stage_kv(t0 + 64, cur ^ 1);  // stream next block async

    // ---- fragments from LDS (ds_load_b128 pairs)
    v16bf kf[8], vf[8];
#pragma unroll
    for (int tt = 0; tt < 4; ++tt) {
      kf[2 * tt]     = load_frag_bf16(&kbuf[cur][tt * 16 + ln][8 * half]);
      kf[2 * tt + 1] = load_frag_bf16(&kbuf[cur][tt * 16 + ln][32 + 8 * half]);
    }
#pragma unroll
    for (int f = 0; f < 4; ++f) {
      vf[2 * f]     = load_frag_bf16(&vbuf[cur][f * 16 + ln][8 * half]);
      vf[2 * f + 1] = load_frag_bf16(&vbuf[cur][f * 16 + ln][32 + 8 * half]);
    }

    // ---- scores: [16 rows] x [64 keys]; already in exp2 domain (Q scaled)
    v8f sc[4];
#pragma unroll
    for (int tt = 0; tt < 4; ++tt) {
#pragma unroll
      for (int j = 0; j < 8; ++j) sc[tt][j] = 0.f;
      sc[tt] = wmma_bf16(qa0, kf[2 * tt], sc[tt]);
      sc[tt] = wmma_bf16(qa1, kf[2 * tt + 1], sc[tt]);
    }

    // ---- online softmax; rows live in 16-lane halves
    float corr[8];
#pragma unroll
    for (int j = 0; j < 8; ++j) {
      float v0 = sc[0][j], v1 = sc[1][j], v2 = sc[2][j], v3 = sc[3][j];
      float mx = fmaxf(fmaxf(v0, v1), fmaxf(v2, v3));
#pragma unroll
      for (int off = 1; off < 16; off <<= 1) mx = fmaxf(mx, __shfl_xor(mx, off));
      float mn = fmaxf(m[j], mx);
      float cr = exp2f(m[j] - mn);
      float p0 = exp2f(v0 - mn), p1 = exp2f(v1 - mn);
      float p2 = exp2f(v2 - mn), p3 = exp2f(v3 - mn);
      float ps = (p0 + p1) + (p2 + p3);
#pragma unroll
      for (int off = 1; off < 16; off <<= 1) ps += __shfl_xor(ps, off);
      l[j] = l[j] * cr + ps;
      m[j] = mn;
      corr[j] = cr;
      const int r = j + 8 * half;
      pstage[wave][r][ln]      = (bf16)p0;
      pstage[wave][r][16 + ln] = (bf16)p1;
      pstage[wave][r][32 + ln] = (bf16)p2;
      pstage[wave][r][48 + ln] = (bf16)p3;
    }
#pragma unroll
    for (int f = 0; f < 4; ++f)
#pragma unroll
      for (int j = 0; j < 8; ++j) o[f][j] *= corr[j];

    __syncthreads();  // pstage stores -> fragment loads

    // ---- P @ V : A = P[16x64] from LDS (2 frags), B = preloaded V frags
    v16bf pa0 = load_frag_bf16(&pstage[wave][ln][0] + 8 * half);
    v16bf pa1 = load_frag_bf16(&pstage[wave][ln][0] + 32 + 8 * half);
#pragma unroll
    for (int f = 0; f < 4; ++f) {
      o[f] = wmma_bf16(pa0, vf[2 * f], o[f]);
      o[f] = wmma_bf16(pa1, vf[2 * f + 1], o[f]);
    }
  }

  // ---- normalize (fast rcp) + store concat-head output
  float inv[8];
#pragma unroll
  for (int j = 0; j < 8; ++j) inv[j] = __builtin_amdgcn_rcpf(l[j]);
#pragma unroll
  for (int f = 0; f < 4; ++f) {
    int d = f * 16 + ln;
#pragma unroll
    for (int j = 0; j < 8; ++j) {
      int s = rowBase + j + 8 * half;
      X[((size_t)b * S_ + s) * E_ + h * D_ + d] = (bf16)(o[f][j] * inv[j]);
    }
  }
}

// ---------------- Kernel 3: output projection ----------------
__global__ __launch_bounds__(256) void outproj_kernel(
    const bf16* __restrict__ Xc, const bf16* __restrict__ WoT,
    const float* __restrict__ bo, float* __restrict__ out) {
  const int wave = threadIdx.x >> 5;
  const int lane = threadIdx.x & 31;
  const int half = lane >> 4;
  const int ln   = lane & 15;
  const int rowBase = blockIdx.x * 128 + wave * 16;
  const int colBase = blockIdx.y * 64;

  v8f c[4];
#pragma unroll
  for (int f = 0; f < 4; ++f)
#pragma unroll
    for (int j = 0; j < 8; ++j) c[f][j] = 0.f;

  const bf16* xrow = Xc + (size_t)(rowBase + ln) * E_;
  const bf16* wbase = WoT + (size_t)(colBase + ln) * E_;
  for (int e = 0; e < E_; e += 32) {
    if (e + 128 < E_) __builtin_prefetch(xrow + e + 128, 0, 0);
    v16bf wf[4];
#pragma unroll
    for (int nt = 0; nt < 4; ++nt)
      wf[nt] = load_frag_bf16(wbase + (size_t)(nt * 16) * E_ + e + 8 * half);
    v16bf a = load_frag_bf16(xrow + e + 8 * half);
#pragma unroll
    for (int nt = 0; nt < 4; ++nt) c[nt] = wmma_bf16(a, wf[nt], c[nt]);
  }

#pragma unroll
  for (int nt = 0; nt < 4; ++nt) {
    int col = colBase + nt * 16 + ln;
    float bv = bo[col];
#pragma unroll
    for (int j = 0; j < 8; ++j) {
      int row = rowBase + j + 8 * half;
      out[(size_t)row * E_ + col] = c[nt][j] + bv;
    }
  }
}

extern "C" void kernel_launch(void* const* d_in, const int* in_sizes, int n_in,
                              void* d_out, int out_size, void* d_ws, size_t ws_size,
                              hipStream_t stream) {
  (void)in_sizes; (void)n_in; (void)out_size; (void)ws_size;
  const float* query = (const float*)d_in[0];
  const float* key   = (const float*)d_in[1];
  const float* value = (const float*)d_in[2];
  const float* Wq    = (const float*)d_in[3];
  const float* bq    = (const float*)d_in[4];
  const float* Wk    = (const float*)d_in[5];
  const float* bk    = (const float*)d_in[6];
  const float* Wv    = (const float*)d_in[7];
  const float* bv    = (const float*)d_in[8];
  const float* Wo    = (const float*)d_in[9];
  const float* bo    = (const float*)d_in[10];
  float* out = (float*)d_out;

  char* ws = (char*)d_ws;
  const size_t WSZ  = (size_t)H_ * D_ * E_ * sizeof(bf16);      // 2 MB
  const size_t QSZ  = (size_t)B_ * H_ * S_ * D_ * sizeof(bf16); // 16 MB
  bf16* WqT = (bf16*)(ws + 0 * WSZ);
  bf16* WkT = (bf16*)(ws + 1 * WSZ);
  bf16* WvT = (bf16*)(ws + 2 * WSZ);
  bf16* WoT = (bf16*)(ws + 3 * WSZ);
  bf16* Qb  = (bf16*)(ws + 4 * WSZ);
  bf16* Kb  = (bf16*)(ws + 4 * WSZ + 1 * QSZ);
  bf16* VTb = (bf16*)(ws + 4 * WSZ + 2 * QSZ);
  bf16* Xb  = (bf16*)(ws + 4 * WSZ + 3 * QSZ);

  const int cvtBlocks = (H_ * E_ * D_ + 255) / 256;
  cvt_transpose_kernel<<<cvtBlocks, 256, 0, stream>>>(Wq, WqT, H_, E_, D_);
  cvt_transpose_kernel<<<cvtBlocks, 256, 0, stream>>>(Wk, WkT, H_, E_, D_);
  cvt_transpose_kernel<<<cvtBlocks, 256, 0, stream>>>(Wv, WvT, H_, E_, D_);
  cvt_transpose_kernel<<<cvtBlocks, 256, 0, stream>>>(Wo, WoT, 1, E_, E_);

  // fold softmax scale * log2(e) into Q so scores land in the exp2 domain
  const float qscale = 0.125f * 1.4426950408889634f;  // (1/sqrt(D)) * log2e
  dim3 pg(S_ / 128, B_ * H_);
  proj_kernel<0><<<pg, 256, 0, stream>>>(query, WqT, bq, Qb, qscale);
  proj_kernel<0><<<pg, 256, 0, stream>>>(key,   WkT, bk, Kb, 1.0f);
  proj_kernel<1><<<pg, 256, 0, stream>>>(value, WvT, bv, VTb, 1.0f);

  attn_kernel<<<dim3(S_ / 64, B_ * H_), 128, 0, stream>>>(Qb, Kb, VTb, Xb);

  outproj_kernel<<<dim3((B_ * S_) / 128, E_ / 64), 256, 0, stream>>>(Xb, WoT, bo, out);
}
